// Block_3341484556448
// MI455X (gfx1250) — compile-verified, matrix operand order checked by reading, not run on hardware
//
#include <hip/hip_runtime.h>
#include <hip/hip_bf16.h>
#include <math.h>

// ---------------------------------------------------------------------------
// GPT block on gfx1250 (MI455X). All matmuls on v_wmma_f32_16x16x32_f16.
// GEMMs and attention K/V use double-buffered LDS tiles filled with
// GLOBAL_LOAD_ASYNC_TO_LDS_B128 (ASYNCcnt); fragments read as 2x b128.
// Flash attention: peeled causal mask, exp2 softmax, DPP-based reductions
// (no ds_bpermute in the hot loop).
// ---------------------------------------------------------------------------

typedef __attribute__((ext_vector_type(16))) _Float16 v16h;
typedef __attribute__((ext_vector_type(8)))  float    v8f;

#define C_DIM   1024
#define N_HEAD  16
#define H_DIM   64
#define B_SZ    4
#define T_SZ    2048
#define DFF     4096
#define M_ROWS  (B_SZ * T_SZ)          // 8192

#define USE_ASYNC_LDS 1

// ISA 7.12.2 f16 A-layout over a K-contiguous 32-element run: dwords
// [4*half..4*half+3] and [8+4*half..11+4*half] -> fragment = two b128 loads.
__device__ __forceinline__ v16h load_frag(const _Float16* base, int half) {
    const uint4* p = reinterpret_cast<const uint4*>(base);
    union { uint4 q[2]; v16h v; } f;
    f.q[0] = p[half];
    f.q[1] = p[2 + half];
    return f.v;
}

__device__ __forceinline__ v8f wmma_f16(v16h a, v16h b, v8f c) {
    return __builtin_amdgcn_wmma_f32_16x16x32_f16(false, a, false, b,
                                                  (short)0, c, false, false);
}

// scale = log2(e)/sqrt(64): softmax becomes exp2 (bare v_exp_f32)
__device__ __forceinline__ v16h scale_frag(v16h a) {
    union { v16h v; _Float16 h[16]; } f; f.v = a;
    #pragma unroll
    for (int i = 0; i < 16; ++i) f.h[i] = f.h[i] * (_Float16)0.18033688f;
    return f.v;
}

__device__ __forceinline__ float gelu_f(float x) {
    const float k = 0.7978845608028654f;
    return 0.5f * x * (1.0f + tanhf(k * (x + 0.044715f * x * x * x)));
}

// ---- DPP lane exchanges (VALU, no LDS pipe) -------------------------------
#define DPP8SEL(a,b,c,d,e,f,g,h) \
    ((a)|((b)<<3)|((c)<<6)|((d)<<9)|((e)<<12)|((f)<<15)|((g)<<18)|((h)<<21))

template <unsigned SEL>
__device__ __forceinline__ float dpp8_f(float x) {
    return __int_as_float(__builtin_amdgcn_mov_dpp8(__float_as_int(x), SEL));
}
__device__ __forceinline__ float ror8_f(float x) {   // xor8 within 16-lane row
    return __int_as_float(__builtin_amdgcn_update_dpp(
        0, __float_as_int(x), 0x128 /*row_ror:8*/, 0xf, 0xf, true));
}
__device__ __forceinline__ float permx16_f(float x) { // swap 16-lane halves
    int i = __float_as_int(x);
    return __int_as_float(__builtin_amdgcn_permlanex16(
        i, i, 0x76543210, 0xfedcba98, false, false));
}
// max/sum over each 16-lane group (attention row reductions)
__device__ __forceinline__ float rowmax16(float x) {
    x = fmaxf(x, dpp8_f<DPP8SEL(1,0,3,2,5,4,7,6)>(x));
    x = fmaxf(x, dpp8_f<DPP8SEL(2,3,0,1,6,7,4,5)>(x));
    x = fmaxf(x, dpp8_f<DPP8SEL(4,5,6,7,0,1,2,3)>(x));
    x = fmaxf(x, ror8_f(x));
    return x;
}
__device__ __forceinline__ float rowsum16(float x) {
    x += dpp8_f<DPP8SEL(1,0,3,2,5,4,7,6)>(x);
    x += dpp8_f<DPP8SEL(2,3,0,1,6,7,4,5)>(x);
    x += dpp8_f<DPP8SEL(4,5,6,7,0,1,2,3)>(x);
    x += ror8_f(x);
    return x;
}
__device__ __forceinline__ float sum32(float x) {     // full-wave sum (LN)
    x += dpp8_f<DPP8SEL(1,0,3,2,5,4,7,6)>(x);
    x += dpp8_f<DPP8SEL(2,3,0,1,6,7,4,5)>(x);
    x += dpp8_f<DPP8SEL(4,5,6,7,0,1,2,3)>(x);
    x += ror8_f(x);
    x += permx16_f(x);
    return x;
}
__device__ __forceinline__ float sum8(float x) {      // sum within 8 lanes
    x += dpp8_f<DPP8SEL(1,0,3,2,5,4,7,6)>(x);
    x += dpp8_f<DPP8SEL(2,3,0,1,6,7,4,5)>(x);
    x += dpp8_f<DPP8SEL(4,5,6,7,0,1,2,3)>(x);
    return x;
}

// ---- async global -> LDS (16B per lane), tracked by ASYNCcnt --------------
__device__ __forceinline__ void async_cp16B(void* lds_dst, const void* gsrc) {
#if USE_ASYNC_LDS
    unsigned loff = (unsigned)(uintptr_t)lds_dst;            // low32 = LDS addr
    unsigned long long ga = (unsigned long long)(uintptr_t)gsrc;
    asm volatile("global_load_async_to_lds_b128 %0, %1, off"
                 :: "v"(loff), "v"(ga) : "memory");
#else
    *reinterpret_cast<uint4*>(lds_dst) = *reinterpret_cast<const uint4*>(gsrc);
#endif
}
__device__ __forceinline__ void async_wait0() {
#if USE_ASYNC_LDS
    asm volatile("s_wait_asynccnt 0" ::: "memory");
#endif
}

// ---------------------------------------------------------------------------
// Weight convert + transpose: [K,N] f32 -> [N,K] f16 (K-major WMMA B operand)
// ---------------------------------------------------------------------------
__global__ __launch_bounds__(256) void cvt_transpose(const float* __restrict__ in,
                                                     _Float16* __restrict__ out,
                                                     int K, int N) {
    __shared__ _Float16 tile[32][33];
    const int k0 = blockIdx.y * 32, n0 = blockIdx.x * 32;
    const int tx = threadIdx.x & 31, ty = threadIdx.x >> 5;
    #pragma unroll
    for (int i = 0; i < 4; ++i)
        tile[ty + i * 8][tx] = (_Float16)in[(size_t)(k0 + ty + i * 8) * N + n0 + tx];
    __syncthreads();
    #pragma unroll
    for (int i = 0; i < 4; ++i)
        out[(size_t)(n0 + ty + i * 8) * K + k0 + tx] = tile[tx][ty + i * 8];
}

// ---------------------------------------------------------------------------
// V transpose per head: qkv v-slice (t,d) -> vT[b,h,d,t]
// ---------------------------------------------------------------------------
__global__ __launch_bounds__(256) void transpose_v(const _Float16* __restrict__ qkv,
                                                   _Float16* __restrict__ vT) {
    __shared__ _Float16 tile[32][33];
    const int bh = blockIdx.z;
    const int t0 = blockIdx.x * 32, d0 = blockIdx.y * 32;
    const int tx = threadIdx.x & 31, ty = threadIdx.x >> 5;
    const _Float16* src = qkv + (size_t)(bh / N_HEAD) * T_SZ * (3 * C_DIM)
                              + 2 * C_DIM + (bh % N_HEAD) * H_DIM;
    _Float16* dst = vT + (size_t)bh * H_DIM * T_SZ;
    #pragma unroll
    for (int i = 0; i < 4; ++i)
        tile[ty + i * 8][tx] = src[(size_t)(t0 + ty + i * 8) * (3 * C_DIM) + d0 + tx];
    __syncthreads();
    #pragma unroll
    for (int i = 0; i < 4; ++i)
        dst[(size_t)(d0 + ty + i * 8) * T_SZ + t0 + tx] = tile[tx][ty + i * 8];
}

// ---------------------------------------------------------------------------
// LayerNorm rows of 1024, fp32 -> f16 (DPP reductions).
// ---------------------------------------------------------------------------
__global__ __launch_bounds__(256) void ln_kernel(const float* __restrict__ x,
                                                 const float* __restrict__ w,
                                                 const float* __restrict__ b,
                                                 _Float16* __restrict__ out) {
    __shared__ float red0[8], red1[8];
    const int row = blockIdx.x;
    const float4 v = reinterpret_cast<const float4*>(x + (size_t)row * C_DIM)[threadIdx.x];
    float s  = sum32(v.x + v.y + v.z + v.w);
    float s2 = sum32(v.x * v.x + v.y * v.y + v.z * v.z + v.w * v.w);
    const int wave = threadIdx.x >> 5, lane = threadIdx.x & 31;
    if (lane == 0) { red0[wave] = s; red1[wave] = s2; }
    __syncthreads();
    if (wave == 0) {
        s  = sum8((lane < 8) ? red0[lane] : 0.f);
        s2 = sum8((lane < 8) ? red1[lane] : 0.f);
        if (lane == 0) {
            float mu = s * (1.0f / C_DIM);
            red0[0] = mu;
            red1[0] = rsqrtf(s2 * (1.0f / C_DIM) - mu * mu + 1e-5f);
        }
    }
    __syncthreads();
    const float mu = red0[0], rstd = red1[0];
    const float4 wv = reinterpret_cast<const float4*>(w)[threadIdx.x];
    const float4 bv = reinterpret_cast<const float4*>(b)[threadIdx.x];
    union { _Float16 h[4]; uint2 u; } o;
    o.h[0] = (_Float16)((v.x - mu) * rstd * wv.x + bv.x);
    o.h[1] = (_Float16)((v.y - mu) * rstd * wv.y + bv.y);
    o.h[2] = (_Float16)((v.z - mu) * rstd * wv.z + bv.z);
    o.h[3] = (_Float16)((v.w - mu) * rstd * wv.w + bv.w);
    reinterpret_cast<uint2*>(out + (size_t)row * C_DIM)[threadIdx.x] = o.u;
}

// ---------------------------------------------------------------------------
// WMMA GEMM with async-LDS double buffering.
// ---------------------------------------------------------------------------
template <int MODE>
__global__ __launch_bounds__(256) void gemm_wmma(const _Float16* __restrict__ A,
                                                 const _Float16* __restrict__ BT,
                                                 const float* __restrict__ bias,
                                                 const float* __restrict__ residual,
                                                 void* __restrict__ out,
                                                 int M, int N, int K) {
    __shared__ _Float16 As[2][128 * 32];
    __shared__ _Float16 Bs[2][128 * 32];
    const int tid  = threadIdx.x;
    const int lane = tid & 31;
    const int wave = tid >> 5;
    const int half = lane >> 4;
    const int l16  = lane & 15;
    const int mbase = blockIdx.y * 128 + (wave & 3) * 32;
    const int nbase = blockIdx.x * 128 + (wave >> 2) * 64;

    const _Float16* Ag = A  + (size_t)(blockIdx.y * 128) * K;
    const _Float16* Bg = BT + (size_t)(blockIdx.x * 128) * K;
    const int srow = tid >> 1;
    const int sseg = (tid & 1) * 16;

    auto stage = [&](int bf, int k0) {
        const _Float16* ga = Ag + (size_t)srow * K + k0 + sseg;
        const _Float16* gb = Bg + (size_t)srow * K + k0 + sseg;
        _Float16* la = &As[bf][srow * 32 + sseg];
        _Float16* lb = &Bs[bf][srow * 32 + sseg];
        async_cp16B(la,     ga);
        async_cp16B(la + 8, ga + 8);
        async_cp16B(lb,     gb);
        async_cp16B(lb + 8, gb + 8);
    };

    stage(0, 0);
    async_wait0();
    __syncthreads();

    v8f acc[2][4] = {};
    int buf = 0;
    for (int k0 = 0; k0 < K; k0 += 32) {
        const bool more = (k0 + 32 < K);
        if (more) stage(buf ^ 1, k0 + 32);

        const _Float16* a_base = &As[buf][((wave & 3) * 32 + l16) * 32];
        const _Float16* b_base = &Bs[buf][((wave >> 2) * 64 + l16) * 32];
        const v16h af0 = load_frag(a_base,           half);
        const v16h af1 = load_frag(a_base + 16 * 32, half);
        const v16h bf0 = load_frag(b_base,           half);
        const v16h bf1 = load_frag(b_base + 16 * 32, half);
        const v16h bf2 = load_frag(b_base + 32 * 32, half);
        const v16h bf3 = load_frag(b_base + 48 * 32, half);
        acc[0][0] = wmma_f16(af0, bf0, acc[0][0]);
        acc[0][1] = wmma_f16(af0, bf1, acc[0][1]);
        acc[0][2] = wmma_f16(af0, bf2, acc[0][2]);
        acc[0][3] = wmma_f16(af0, bf3, acc[0][3]);
        acc[1][0] = wmma_f16(af1, bf0, acc[1][0]);
        acc[1][1] = wmma_f16(af1, bf1, acc[1][1]);
        acc[1][2] = wmma_f16(af1, bf2, acc[1][2]);
        acc[1][3] = wmma_f16(af1, bf3, acc[1][3]);

        if (more) async_wait0();
        __syncthreads();
        buf ^= 1;
    }

    _Float16* o16 = (_Float16*)out;
    float*    o32 = (float*)out;
    #pragma unroll
    for (int mt = 0; mt < 2; ++mt)
        #pragma unroll
        for (int nt = 0; nt < 4; ++nt) {
            const int col = nbase + nt * 16 + l16;
            const float bv = bias[col];
            #pragma unroll
            for (int r = 0; r < 8; ++r) {
                const int row = mbase + mt * 16 + r + 8 * half;
                const size_t idx = (size_t)row * N + col;
                const float v = acc[mt][nt][r] + bv;
                if (MODE == 0)      o16[idx] = (_Float16)v;
                else if (MODE == 1) o32[idx] = residual[idx] + v;
                else                o16[idx] = (_Float16)gelu_f(v);
            }
        }
}

// ---------------------------------------------------------------------------
// Flash attention with cooperative async-LDS K/V staging.
// Block = 128 thr (4 waves), q-range 64 rows, 64-key tiles.
// ktiles == blockIdx.x+1 for ALL 4 waves -> block barriers legal.
// Only the last tile touches the diagonal -> masks peeled out of main loop.
// ---------------------------------------------------------------------------
__global__ __launch_bounds__(128) void attn_kernel(const _Float16* __restrict__ qkv,
                                                   const _Float16* __restrict__ vT,
                                                   _Float16* __restrict__ y) {
    __shared__ _Float16 Ks[2][64 * 64];     // [key_local][d]
    __shared__ _Float16 Vs[2][64 * 64];     // [d][key_local]
    __shared__ _Float16 Plds[4][16 * 64];
    const int tid  = threadIdx.x;
    const int lane = tid & 31;
    const int wave = tid >> 5;
    const int half = lane >> 4;
    const int l16  = lane & 15;
    const int bh = blockIdx.y;
    const int b  = bh / N_HEAD;
    const int h  = bh % N_HEAD;
    const int q0 = blockIdx.x * 64 + wave * 16;

    const size_t rs = 3 * C_DIM;
    const _Float16* qp  = qkv + (size_t)b * T_SZ * rs + h * H_DIM;
    const _Float16* kp  = qp + C_DIM;
    const _Float16* vTp = vT + (size_t)bh * H_DIM * T_SZ;

    const _Float16* qrow = qp + (size_t)(q0 + l16) * rs;
    const v16h qf0 = scale_frag(load_frag(qrow,      half));  // log2e/sqrt(hd)
    const v16h qf1 = scale_frag(load_frag(qrow + 32, half));

    const int srow = tid >> 1;            // 0..63
    const int sseg = (tid & 1) * 32;      // 32 f16 = 64B per thread per tile
    auto stageKV = [&](int bf, int kbase) {
        const _Float16* gk = kp  + (size_t)(kbase + srow) * rs + sseg;
        const _Float16* gv = vTp + (size_t)srow * T_SZ + kbase + sseg;
        _Float16* lk = &Ks[bf][srow * 64 + sseg];
        _Float16* lv = &Vs[bf][srow * 64 + sseg];
        async_cp16B(lk,      gk);
        async_cp16B(lk + 8,  gk + 8);
        async_cp16B(lk + 16, gk + 16);
        async_cp16B(lk + 24, gk + 24);
        async_cp16B(lv,      gv);
        async_cp16B(lv + 8,  gv + 8);
        async_cp16B(lv + 16, gv + 16);
        async_cp16B(lv + 24, gv + 24);
    };

    float mi[8], li[8];
    #pragma unroll
    for (int r = 0; r < 8; ++r) { mi[r] = -3.0e30f; li[r] = 0.f; }
    v8f o0 = {}, o1 = {}, o2 = {}, o3 = {};

    const int ktiles = blockIdx.x + 1;    // uniform across the block
    stageKV(0, 0);
    async_wait0();
    __syncthreads();

    int buf = 0;
    auto tile_body = [&](int kt, bool masked) {
        const int kbase = kt * 64;
        const bool more = (kt + 1 < ktiles);
        if (more) stageKV(buf ^ 1, kbase + 64);

        // ----- S = Q * K^T (8 wmma, fragments from LDS) -----
        const _Float16* kb = &Ks[buf][l16 * 64];
        v8f sA = {}, sB = {}, sC = {}, sD = {};
        sA = wmma_f16(qf0, load_frag(kb,                half), sA);
        sA = wmma_f16(qf1, load_frag(kb + 32,           half), sA);
        sB = wmma_f16(qf0, load_frag(kb + 16 * 64,      half), sB);
        sB = wmma_f16(qf1, load_frag(kb + 16 * 64 + 32, half), sB);
        sC = wmma_f16(qf0, load_frag(kb + 32 * 64,      half), sC);
        sC = wmma_f16(qf1, load_frag(kb + 32 * 64 + 32, half), sC);
        sD = wmma_f16(qf0, load_frag(kb + 48 * 64,      half), sD);
        sD = wmma_f16(qf1, load_frag(kb + 48 * 64 + 32, half), sD);

        // ----- online softmax (base-2, DPP reductions) -----
        #pragma unroll
        for (int r = 0; r < 8; ++r) {
            float s0 = sA[r], s1 = sB[r], s2 = sC[r], s3 = sD[r];
            if (masked) {
                const int qm = q0 + r + 8 * half;
                if (kbase + l16 > qm)      s0 = -3.0e30f;
                if (kbase + 16 + l16 > qm) s1 = -3.0e30f;
                if (kbase + 32 + l16 > qm) s2 = -3.0e30f;
                if (kbase + 48 + l16 > qm) s3 = -3.0e30f;
            }
            const float m  = rowmax16(fmaxf(fmaxf(s0, s1), fmaxf(s2, s3)));
            const float mn = fmaxf(mi[r], m);
            const float alpha = exp2f(mi[r] - mn);
            const float p0 = exp2f(s0 - mn);
            const float p1 = exp2f(s1 - mn);
            const float p2 = exp2f(s2 - mn);
            const float p3 = exp2f(s3 - mn);
            const float ps = rowsum16((p0 + p1) + (p2 + p3));
            li[r] = li[r] * alpha + ps;
            mi[r] = mn;
            o0[r] *= alpha; o1[r] *= alpha; o2[r] *= alpha; o3[r] *= alpha;
            const int mloc = (r + 8 * half) * 64;
            Plds[wave][mloc + l16]      = (_Float16)p0;
            Plds[wave][mloc + 16 + l16] = (_Float16)p1;
            Plds[wave][mloc + 32 + l16] = (_Float16)p2;
            Plds[wave][mloc + 48 + l16] = (_Float16)p3;
        }

        // wave-internal LDS ordering (DS in-order within wave)
        __builtin_amdgcn_fence(__ATOMIC_ACQ_REL, "wavefront");
        __builtin_amdgcn_wave_barrier();
        const v16h pf0 = load_frag(&Plds[wave][l16 * 64],      half);
        const v16h pf1 = load_frag(&Plds[wave][l16 * 64 + 32], half);
        __builtin_amdgcn_wave_barrier();
        __builtin_amdgcn_fence(__ATOMIC_ACQ_REL, "wavefront");

        // ----- O += P * V (8 wmma, V from LDS) -----
        const _Float16* vbld = &Vs[buf][l16 * 64];
        o0 = wmma_f16(pf0, load_frag(vbld,                half), o0);
        o0 = wmma_f16(pf1, load_frag(vbld + 32,           half), o0);
        o1 = wmma_f16(pf0, load_frag(vbld + 16 * 64,      half), o1);
        o1 = wmma_f16(pf1, load_frag(vbld + 16 * 64 + 32, half), o1);
        o2 = wmma_f16(pf0, load_frag(vbld + 32 * 64,      half), o2);
        o2 = wmma_f16(pf1, load_frag(vbld + 32 * 64 + 32, half), o2);
        o3 = wmma_f16(pf0, load_frag(vbld + 48 * 64,      half), o3);
        o3 = wmma_f16(pf1, load_frag(vbld + 48 * 64 + 32, half), o3);

        if (more) async_wait0();
        __syncthreads();
        buf ^= 1;
    };

    for (int kt = 0; kt < ktiles - 1; ++kt) tile_body(kt, false);  // mask-free
    tile_body(ktiles - 1, true);                                    // diagonal

#define OUT_STEP(ONT, NT)                                                    \
    {                                                                        \
        const int dcol = (NT) * 16 + l16;                                    \
        _Pragma("unroll")                                                    \
        for (int r = 0; r < 8; ++r) {                                        \
            const int t = q0 + r + 8 * half;                                 \
            y[((size_t)b * T_SZ + t) * C_DIM + h * H_DIM + dcol] =           \
                (_Float16)(ONT[r] / li[r]);                                  \
        }                                                                    \
    }
    OUT_STEP(o0, 0) OUT_STEP(o1, 1) OUT_STEP(o2, 2) OUT_STEP(o3, 3)
#undef OUT_STEP
}

// ---------------------------------------------------------------------------
// Host-side orchestration
// ---------------------------------------------------------------------------
static inline char* carve(char*& p, size_t bytes) {
    char* r = p;
    p += (bytes + 255) & ~size_t(255);
    return r;
}

extern "C" void kernel_launch(void* const* d_in, const int* in_sizes, int n_in,
                              void* d_out, int out_size, void* d_ws, size_t ws_size,
                              hipStream_t stream) {
    const float* x      = (const float*)d_in[0];
    const float* ln1_w  = (const float*)d_in[1];
    const float* ln1_b  = (const float*)d_in[2];
    const float* w_attn = (const float*)d_in[3];
    const float* b_attn = (const float*)d_in[4];
    const float* w_proj = (const float*)d_in[5];
    const float* b_proj = (const float*)d_in[6];
    const float* ln2_w  = (const float*)d_in[7];
    const float* ln2_b  = (const float*)d_in[8];
    const float* w_fc   = (const float*)d_in[9];
    const float* b_fc   = (const float*)d_in[10];
    const float* w_fc2  = (const float*)d_in[11];
    const float* b_fc2  = (const float*)d_in[12];

    const int M = M_ROWS, C = C_DIM;

    char* p = (char*)d_ws;
    _Float16* hf16  = (_Float16*)carve(p, (size_t)M * C * 2);
    _Float16* watT  = (_Float16*)carve(p, (size_t)3 * C * C * 2);
    _Float16* wprT  = (_Float16*)carve(p, (size_t)C * C * 2);
    _Float16* wfcT  = (_Float16*)carve(p, (size_t)DFF * C * 2);
    _Float16* wf2T  = (_Float16*)carve(p, (size_t)C * DFF * 2);
    _Float16* qkv16 = (_Float16*)carve(p, (size_t)M * 3 * C * 2);
    _Float16* vT16  = (_Float16*)carve(p, (size_t)M * C * 2);
    _Float16* y16   = (_Float16*)carve(p, (size_t)M * C * 2);
    float*    x2    = (float*)   carve(p, (size_t)M * C * 4);
    _Float16* h2f16 = (_Float16*)carve(p, (size_t)M * C * 2);
    _Float16* g16   = (_Float16*)carve(p, (size_t)M * DFF * 2);

    cvt_transpose<<<dim3(3 * C / 32, C / 32), 256, 0, stream>>>(w_attn, watT, C, 3 * C);
    cvt_transpose<<<dim3(C / 32, C / 32),     256, 0, stream>>>(w_proj, wprT, C, C);
    cvt_transpose<<<dim3(DFF / 32, C / 32),   256, 0, stream>>>(w_fc,  wfcT, C, DFF);
    cvt_transpose<<<dim3(C / 32, DFF / 32),   256, 0, stream>>>(w_fc2, wf2T, DFF, C);

    ln_kernel<<<M, 256, 0, stream>>>(x, ln1_w, ln1_b, hf16);
    gemm_wmma<0><<<dim3(3 * C / 128, M / 128), 256, 0, stream>>>(
        hf16, watT, b_attn, nullptr, qkv16, M, 3 * C, C);
    transpose_v<<<dim3(T_SZ / 32, H_DIM / 32, B_SZ * N_HEAD), 256, 0, stream>>>(qkv16, vT16);
    attn_kernel<<<dim3(T_SZ / 64, B_SZ * N_HEAD), 128, 0, stream>>>(qkv16, vT16, y16);
    gemm_wmma<1><<<dim3(C / 128, M / 128), 256, 0, stream>>>(
        y16, wprT, b_proj, x, x2, M, C, C);
    ln_kernel<<<M, 256, 0, stream>>>(x2, ln2_w, ln2_b, h2f16);
    gemm_wmma<2><<<dim3(DFF / 128, M / 128), 256, 0, stream>>>(
        h2f16, wfcT, b_fc, nullptr, g16, M, DFF, C);
    gemm_wmma<1><<<dim3(C / 128, M / 128), 256, 0, stream>>>(
        g16, wf2T, b_fc2, x2, d_out, M, C, DFF);
}
